// Zamba2MambaDecoderLayer_49718541418503
// MI455X (gfx1250) — compile-verified, matrix operand order checked by reading, not run on hardware
//
#include <hip/hip_runtime.h>

// ---------------- model constants ----------------
constexpr int HDIM  = 2048;            // hidden
constexpr int IDIM  = 4096;            // intermediate
constexpr int NST   = 64;              // ssm state
constexpr int NH    = 64;              // heads
constexpr int PD    = 64;              // head dim
constexpr int CHK   = 256;             // chunk
constexpr int NCHK  = 8;               // 2048 / 256
constexpr int CONVD = IDIM + 2 * NST;  // 4224
constexpr int PROJD = IDIM + CONVD + NH; // 8384
constexpr int BATCH = 2;
constexpr int TLEN  = 2048;
constexpr int MTOK  = BATCH * TLEN;    // 4096 rows

typedef __bf16 bf16;
typedef __attribute__((ext_vector_type(8)))  bf16  v8bf;
typedef __attribute__((ext_vector_type(16))) bf16  v16bf;
typedef __attribute__((ext_vector_type(8)))  float v8f;

// ---------------- small helpers ----------------
static __device__ inline bf16 f2bf(float f) {
  union { float f; unsigned int u; } cv{f};
  unsigned int u = cv.u;
  u = (u + 0x7FFFu + ((u >> 16) & 1u)) >> 16;   // round-to-nearest-even
  union { unsigned short s; bf16 b; } o{(unsigned short)u};
  return o.b;
}
static __device__ inline float bf2f(bf16 b) {
  union { unsigned short s; bf16 b; } i{0};
  i.b = b;
  union { unsigned int u; float f; } o{(unsigned int)i.s << 16};
  return o.f;
}
static __device__ inline float softplusf(float z) {
  return (z > 20.f) ? z : log1pf(expf(z));
}
static __device__ inline float siluf(float v) {
  return v / (1.f + expf(-v));
}

static __device__ inline v8f wmma_bf16(v16bf a, v16bf b, v8f c) {
  // D = A(16x32) * B(32x16) + C ; fp32 accumulate
  return __builtin_amdgcn_wmma_f32_16x16x32_bf16(
      false, a, false, b, (short)0, c, false, false);
}

// A/B fragment loader for sources stored row-major over k:
//   lane index dim r (A: output row, B: output column with src[n][k])
//   element j -> k = k0 + (j>=8 ? 16 : 0) + (lane>=16 ? 8 : 0) + (j&7)
static __device__ inline v16bf frag_ld(const bf16* base, int ld, int r, int k0,
                                       int lane) {
  const bf16* p = base + (size_t)r * ld + k0 + ((lane >> 4) << 3);
  v8bf lo = *(const v8bf*)(p);
  v8bf hi = *(const v8bf*)(p + 16);
  v16bf f;
#pragma unroll
  for (int j = 0; j < 8; ++j) { f[j] = lo[j]; f[j + 8] = hi[j]; }
  return f;
}
static __device__ inline v16bf frag_ld_f32(const float* base, int ld, int r,
                                           int k0, int lane) {
  const float* p = base + (size_t)r * ld + k0 + ((lane >> 4) << 3);
  v16bf f;
#pragma unroll
  for (int j = 0; j < 8; ++j) { f[j] = f2bf(p[j]); f[j + 8] = f2bf(p[j + 16]); }
  return f;
}

// ---------------- kernels ----------------

__global__ void cvt_bf16_kernel(const float* __restrict__ in,
                                bf16* __restrict__ out, long n) {
  long i = (long)blockIdx.x * blockDim.x + threadIdx.x;
  long st = (long)gridDim.x * blockDim.x;
  for (; i < n; i += st) out[i] = f2bf(in[i]);
}

__global__ __launch_bounds__(256)
void rmsnorm_in_kernel(const float* __restrict__ x, const float* __restrict__ w,
                       bf16* __restrict__ xn) {
  __shared__ float red[256];
  int row = blockIdx.x, tid = threadIdx.x;
  const float* xr = x + (size_t)row * HDIM;
  float ss = 0.f;
  for (int i = tid; i < HDIM; i += 256) { float v = xr[i]; ss += v * v; }
  red[tid] = ss; __syncthreads();
  for (int o = 128; o > 0; o >>= 1) {
    if (tid < o) red[tid] += red[tid + o];
    __syncthreads();
  }
  float rs = rsqrtf(red[0] / (float)HDIM + 1e-5f);
  for (int i = tid; i < HDIM; i += 256)
    xn[(size_t)row * HDIM + i] = f2bf(xr[i] * rs * w[i]);
}

// C[m,n] = sum_k A[m,k] * W[n,k]  (+ residual), bf16 inputs, fp32 out.
// grid = (N/64, M/256); each wave owns a 32x64 tile:
//   per k-step: 2 A frags + 4 B frags -> 8 WMMAs (~21 FLOP/byte from cache)
__global__ __launch_bounds__(256)
void gemm_wave_kernel(const bf16* __restrict__ A, int lda,
                      const bf16* __restrict__ W, int ldw,
                      float* __restrict__ C, int ldc, int K,
                      const float* __restrict__ residual) {
  int tid = threadIdx.x, lane = tid & 31, wv = tid >> 5;
  int m0 = blockIdx.y * 256 + wv * 32;
  int n0 = blockIdx.x * 64;
  int mr = m0 + (lane & 15);
  int nc = n0 + (lane & 15);
  v8f acc[2][4];
#pragma unroll
  for (int s = 0; s < 2; ++s)
#pragma unroll
    for (int c = 0; c < 4; ++c) acc[s][c] = (v8f){};

  for (int k0 = 0; k0 < K; k0 += 32) {
    __builtin_prefetch(A + (size_t)mr * lda + k0 + 256, 0, 1);
    __builtin_prefetch(W + (size_t)nc * ldw + k0 + 256, 0, 1);
    v16bf a0 = frag_ld(A, lda, mr,      k0, lane);
    v16bf a1 = frag_ld(A, lda, mr + 16, k0, lane);
    v16bf b0 = frag_ld(W, ldw, nc +  0, k0, lane);
    v16bf b1 = frag_ld(W, ldw, nc + 16, k0, lane);
    v16bf b2 = frag_ld(W, ldw, nc + 32, k0, lane);
    v16bf b3 = frag_ld(W, ldw, nc + 48, k0, lane);
    acc[0][0] = wmma_bf16(a0, b0, acc[0][0]);
    acc[0][1] = wmma_bf16(a0, b1, acc[0][1]);
    acc[0][2] = wmma_bf16(a0, b2, acc[0][2]);
    acc[0][3] = wmma_bf16(a0, b3, acc[0][3]);
    acc[1][0] = wmma_bf16(a1, b0, acc[1][0]);
    acc[1][1] = wmma_bf16(a1, b1, acc[1][1]);
    acc[1][2] = wmma_bf16(a1, b2, acc[1][2]);
    acc[1][3] = wmma_bf16(a1, b3, acc[1][3]);
  }
  int hf8 = (lane >> 4) << 3;
  int nb = n0 + (lane & 15);
#pragma unroll
  for (int s = 0; s < 2; ++s) {
#pragma unroll
    for (int r = 0; r < 8; ++r) {
      int m = m0 + s * 16 + r + hf8;
      float* crow = C + (size_t)m * ldc;
      float r0 = 0.f, r1 = 0.f, r2 = 0.f, r3 = 0.f;
      if (residual) {
        const float* rr = residual + (size_t)m * ldc;
        r0 = rr[nb]; r1 = rr[nb + 16]; r2 = rr[nb + 32]; r3 = rr[nb + 48];
      }
      crow[nb]      = acc[s][0][r] + r0;
      crow[nb + 16] = acc[s][1][r] + r1;
      crow[nb + 32] = acc[s][2][r] + r2;
      crow[nb + 48] = acc[s][3][r] + r3;
    }
  }
}

// causal depthwise conv (K=4) + bias + SiLU over the xBC slice of proj
__global__ void conv_silu_kernel(const float* __restrict__ proj,
                                 const float* __restrict__ cw,
                                 const float* __restrict__ cb,
                                 float* __restrict__ xbc) {
  int chn = blockIdx.x * 256 + threadIdx.x;
  if (chn >= CONVD) return;
  int t = blockIdx.y, b = blockIdx.z;
  float s = cb[chn];
  const float* col = proj + (size_t)(b * TLEN) * PROJD + IDIM + chn;
#pragma unroll
  for (int k = 0; k < 4; ++k) {
    int tt = t - 3 + k;
    if (tt >= 0) s += cw[chn * 4 + k] * col[(size_t)tt * PROJD];
  }
  xbc[(size_t)(b * TLEN + t) * CONVD + chn] = siluf(s);
}

// Per (b, chunk, head): dt prep + cumsum, S = mask(C B^T), Y_diag = S X,
// chunk states = (X*decay)^T B.  Everything WMMA out of LDS (~290 KB/WG).
__global__ __launch_bounds__(256)
void ssd_chunk_kernel(const float* __restrict__ xbc,
                      const float* __restrict__ proj,
                      const float* __restrict__ dt_bias,
                      const float* __restrict__ A_log,
                      float* __restrict__ Y, float* __restrict__ acum_g,
                      float* __restrict__ states_g,
                      float* __restrict__ cdec_g) {
  extern __shared__ char smem[];
  bf16* sX  = (bf16*)smem;        // [256][64]  X * dt
  bf16* sXt = sX  + CHK * PD;     // [64][256]
  bf16* sB  = sXt + CHK * PD;     // [256][64]
  bf16* sBt = sB  + CHK * PD;     // [64][256]
  bf16* sC  = sBt + CHK * PD;     // [256][64]
  bf16* sS  = sC  + CHK * PD;     // [256][256] masked decay * (C B^T)
  float* sAc = (float*)(sS + CHK * CHK);  // [256] inclusive cumsum(dA)
  float* sDs = sAc + CHK;                 // [256] exp(Ac_last - Ac[l])

  int tid = threadIdx.x, lane = tid & 31, wv = tid >> 5;
  int h = blockIdx.x, ch = blockIdx.y, b = blockIdx.z;
  int rowbase = b * TLEN + ch * CHK;

  // ---- stage 1: load + dt prep (thread == chunk row); float4 vectorized ----
  {
    int l = tid;
    int row = rowbase + l;
    float z = proj[(size_t)row * PROJD + IDIM + CONVD + h] + dt_bias[h];
    float dts = softplusf(z);
    sAc[l] = dts * (-expf(A_log[h]));  // dA
    const float* xr = xbc + (size_t)row * CONVD;
    const float4* x4 = (const float4*)(xr + h * PD);
    const float4* b4 = (const float4*)(xr + IDIM);
    const float4* c4 = (const float4*)(xr + IDIM + NST);
#pragma unroll
    for (int q = 0; q < 8; ++q) {           // 8 bf16 per iter
      float4 v0 = x4[q * 2], v1 = x4[q * 2 + 1];
      v8bf vx = {f2bf(v0.x * dts), f2bf(v0.y * dts), f2bf(v0.z * dts),
                 f2bf(v0.w * dts), f2bf(v1.x * dts), f2bf(v1.y * dts),
                 f2bf(v1.z * dts), f2bf(v1.w * dts)};
      *(v8bf*)(sX + l * PD + q * 8) = vx;
#pragma unroll
      for (int e = 0; e < 8; ++e) sXt[(q * 8 + e) * CHK + l] = vx[e];

      float4 w0 = b4[q * 2], w1 = b4[q * 2 + 1];
      v8bf vb = {f2bf(w0.x), f2bf(w0.y), f2bf(w0.z), f2bf(w0.w),
                 f2bf(w1.x), f2bf(w1.y), f2bf(w1.z), f2bf(w1.w)};
      *(v8bf*)(sB + l * NST + q * 8) = vb;
#pragma unroll
      for (int e = 0; e < 8; ++e) sBt[(q * 8 + e) * CHK + l] = vb[e];

      float4 u0 = c4[q * 2], u1 = c4[q * 2 + 1];
      v8bf vc = {f2bf(u0.x), f2bf(u0.y), f2bf(u0.z), f2bf(u0.w),
                 f2bf(u1.x), f2bf(u1.y), f2bf(u1.z), f2bf(u1.w)};
      *(v8bf*)(sC + l * NST + q * 8) = vc;
    }
  }
  __syncthreads();
  // inclusive Hillis-Steele scan of dA -> A_cum
  for (int off = 1; off < CHK; off <<= 1) {
    float v = sAc[tid];
    float add = (tid >= off) ? sAc[tid - off] : 0.f;
    __syncthreads();
    sAc[tid] = v + add;
    __syncthreads();
  }
  float aclast = sAc[CHK - 1];
  sDs[tid] = expf(aclast - sAc[tid]);
  acum_g[(((size_t)b * NH + h) * NCHK + ch) * CHK + tid] = sAc[tid];
  if (tid == 0) cdec_g[((size_t)b * NH + h) * NCHK + ch] = expf(aclast);
  __syncthreads();

  // ---- stage 2: S[i][j] = (i>=j) ? exp(Ac[i]-Ac[j]) * sum_n C[i,n]B[j,n] : 0
  for (int tt = wv * 32; tt < wv * 32 + 32; ++tt) {
    int i0 = (tt >> 4) << 4, j0 = (tt & 15) << 4;
    int jj = j0 + (lane & 15);
    int hf8 = (lane >> 4) << 3;
    if (i0 + 15 < j0) {  // fully above diagonal (wave-uniform branch)
#pragma unroll
      for (int r = 0; r < 8; ++r) sS[(i0 + r + hf8) * CHK + jj] = f2bf(0.f);
      continue;
    }
    v8f acc = {};
    acc = wmma_bf16(frag_ld(sC, NST, i0 + (lane & 15), 0, lane),
                    frag_ld(sB, NST, jj, 0, lane), acc);
    acc = wmma_bf16(frag_ld(sC, NST, i0 + (lane & 15), 32, lane),
                    frag_ld(sB, NST, jj, 32, lane), acc);
#pragma unroll
    for (int r = 0; r < 8; ++r) {
      int i = i0 + r + hf8;
      float v = (i >= jj) ? acc[r] * expf(sAc[i] - sAc[jj]) : 0.f;
      sS[i * CHK + jj] = f2bf(v);
    }
  }
  __syncthreads();

  // ---- stage 3: Y_diag = S @ Xdt ; wave owns rows [wv*32, wv*32+32)
  for (int s = 0; s < 2; ++s) {
    int i0 = wv * 32 + s * 16;
    int ir = i0 + (lane & 15);
    int pc = lane & 15;
    v8f y0 = {}, y1 = {}, y2 = {}, y3 = {};
    for (int k0 = 0; k0 < CHK; k0 += 32) {
      v16bf a = frag_ld(sS, CHK, ir, k0, lane);
      y0 = wmma_bf16(a, frag_ld(sXt, CHK, pc +  0, k0, lane), y0);
      y1 = wmma_bf16(a, frag_ld(sXt, CHK, pc + 16, k0, lane), y1);
      y2 = wmma_bf16(a, frag_ld(sXt, CHK, pc + 32, k0, lane), y2);
      y3 = wmma_bf16(a, frag_ld(sXt, CHK, pc + 48, k0, lane), y3);
    }
    int hf8 = (lane >> 4) << 3;
#pragma unroll
    for (int r = 0; r < 8; ++r) {
      int m = i0 + r + hf8;
      float* yrow = Y + (size_t)(rowbase + m) * IDIM + h * PD;
      yrow[pc] = y0[r]; yrow[pc + 16] = y1[r];
      yrow[pc + 32] = y2[r]; yrow[pc + 48] = y3[r];
    }
  }

  // ---- stage 4: states[p][n] = sum_l Xdt[l,p]*decay[l]*B[l,n]
  for (int t2 = 0; t2 < 2; ++t2) {
    int tt = wv * 2 + t2;                 // 16 tiles / 8 waves
    int p0 = (tt >> 2) << 4, n0 = (tt & 3) << 4;
    int pr = p0 + (lane & 15);
    int nc = n0 + (lane & 15);
    v8f acc = {};
    for (int k0 = 0; k0 < CHK; k0 += 32) {
      int kb = k0 + ((lane >> 4) << 3);
      v16bf a;
#pragma unroll
      for (int j = 0; j < 8; ++j) {
        int k1 = kb + j, k2 = kb + 16 + j;
        a[j]     = f2bf(bf2f(sXt[pr * CHK + k1]) * sDs[k1]);
        a[j + 8] = f2bf(bf2f(sXt[pr * CHK + k2]) * sDs[k2]);
      }
      acc = wmma_bf16(a, frag_ld(sBt, CHK, nc, k0, lane), acc);
    }
    float* sg = states_g + (((size_t)b * NCHK + ch) * NH + h) * (PD * NST);
    int hf8 = (lane >> 4) << 3;
#pragma unroll
    for (int r = 0; r < 8; ++r) sg[(p0 + r + hf8) * NST + nc] = acc[r];
  }
}

// sequential inter-chunk recurrence (emits state BEFORE each chunk)
__global__ __launch_bounds__(256)
void ssd_scan_kernel(const float* __restrict__ states,
                     const float* __restrict__ cdec,
                     float* __restrict__ prev) {
  int h = blockIdx.x, b = blockIdx.y, tid = threadIdx.x;
  float carry[16];
#pragma unroll
  for (int e = 0; e < 16; ++e) carry[e] = 0.f;
  for (int c = 0; c < NCHK; ++c) {
    size_t off = (((size_t)b * NCHK + c) * NH + h) * (PD * NST) + tid * 16;
    float dec = cdec[((size_t)b * NH + h) * NCHK + c];
#pragma unroll
    for (int e = 0; e < 16; ++e) {
      prev[off + e] = carry[e];
      carry[e] = dec * carry[e] + states[off + e];
    }
  }
}

// Y += (C * exp(Acum)) @ prev^T + D[h] * x
__global__ __launch_bounds__(256)
void ssd_off_kernel(const float* __restrict__ xbc,
                    const float* __restrict__ acum_g,
                    const float* __restrict__ prev,
                    const float* __restrict__ Dp, float* __restrict__ Y) {
  int tid = threadIdx.x, lane = tid & 31, wv = tid >> 5;
  int h = blockIdx.x, ch = blockIdx.y, b = blockIdx.z;
  int rowbase = b * TLEN + ch * CHK;
  const float* ac = acum_g + (((size_t)b * NH + h) * NCHK + ch) * CHK;
  const float* pv = prev + (((size_t)b * NCHK + ch) * NH + h) * (PD * NST);
  float Dh = Dp[h];
  for (int s = 0; s < 2; ++s) {
    int i0 = wv * 32 + s * 16;
    int ir = i0 + (lane & 15);
    int pc = lane & 15;
    float sd = expf(ac[ir]);
    v8f y0 = {}, y1 = {}, y2 = {}, y3 = {};
    for (int k0 = 0; k0 < NST; k0 += 32) {
      int kb = k0 + ((lane >> 4) << 3);
      const float* cr =
          xbc + (size_t)(rowbase + ir) * CONVD + IDIM + NST + kb;
      v16bf a;
#pragma unroll
      for (int j = 0; j < 8; ++j) {
        a[j] = f2bf(cr[j] * sd);
        a[j + 8] = f2bf(cr[16 + j] * sd);
      }
      y0 = wmma_bf16(a, frag_ld_f32(pv, NST, pc +  0, k0, lane), y0);
      y1 = wmma_bf16(a, frag_ld_f32(pv, NST, pc + 16, k0, lane), y1);
      y2 = wmma_bf16(a, frag_ld_f32(pv, NST, pc + 32, k0, lane), y2);
      y3 = wmma_bf16(a, frag_ld_f32(pv, NST, pc + 48, k0, lane), y3);
    }
    int hf8 = (lane >> 4) << 3;
#pragma unroll
    for (int r = 0; r < 8; ++r) {
      int m = i0 + r + hf8;
      const float* xr = xbc + (size_t)(rowbase + m) * CONVD + h * PD;
      float* yrow = Y + (size_t)(rowbase + m) * IDIM + h * PD;
      yrow[pc]      += y0[r] + Dh * xr[pc];
      yrow[pc + 16] += y1[r] + Dh * xr[pc + 16];
      yrow[pc + 32] += y2[r] + Dh * xr[pc + 32];
      yrow[pc + 48] += y3[r] + Dh * xr[pc + 48];
    }
  }
}

// y = Y * silu(gate); RMSNorm; -> bf16
__global__ __launch_bounds__(256)
void gated_norm_kernel(float* __restrict__ Y, const float* __restrict__ proj,
                       const float* __restrict__ w, bf16* __restrict__ yb) {
  __shared__ float red[256];
  int row = blockIdx.x, tid = threadIdx.x;
  float* yr = Y + (size_t)row * IDIM;
  const float* gr = proj + (size_t)row * PROJD;  // gate = first IDIM cols
  float ss = 0.f;
  for (int i = tid; i < IDIM; i += 256) {
    float v = yr[i] * siluf(gr[i]);
    yr[i] = v;
    ss += v * v;
  }
  red[tid] = ss; __syncthreads();
  for (int o = 128; o > 0; o >>= 1) {
    if (tid < o) red[tid] += red[tid + o];
    __syncthreads();
  }
  float rs = rsqrtf(red[0] / (float)IDIM + 1e-5f);
  for (int i = tid; i < IDIM; i += 256)
    yb[(size_t)row * IDIM + i] = f2bf(yr[i] * rs * w[i]);
}

// ---------------- launcher ----------------
extern "C" void kernel_launch(void* const* d_in, const int* in_sizes, int n_in,
                              void* d_out, int out_size, void* d_ws,
                              size_t ws_size, hipStream_t stream) {
  (void)in_sizes; (void)n_in; (void)out_size; (void)ws_size;
  const float* hs   = (const float*)d_in[0];
  const float* lnw  = (const float*)d_in[1];
  const float* win  = (const float*)d_in[2];
  const float* cw   = (const float*)d_in[3];
  const float* cb   = (const float*)d_in[4];
  const float* dtb  = (const float*)d_in[5];
  const float* alog = (const float*)d_in[6];
  const float* Dp   = (const float*)d_in[7];
  const float* nw   = (const float*)d_in[8];
  const float* wout = (const float*)d_in[9];
  float* out = (float*)d_out;

  char* ws = (char*)d_ws;
  size_t off = 0;
  auto alloc = [&](size_t bytes) -> char* {
    char* p = ws + off;
    off += (bytes + 255) & ~(size_t)255;
    return p;
  };
  bf16*  w_in_b  = (bf16*)alloc((size_t)PROJD * HDIM * sizeof(bf16));
  bf16*  w_out_b = (bf16*)alloc((size_t)HDIM * IDIM * sizeof(bf16));
  bf16*  xn_b    = (bf16*)alloc((size_t)MTOK * HDIM * sizeof(bf16));
  float* proj    = (float*)alloc((size_t)MTOK * PROJD * sizeof(float));
  float* xbc     = (float*)alloc((size_t)MTOK * CONVD * sizeof(float));
  float* acum    = (float*)alloc((size_t)BATCH * NH * NCHK * CHK * sizeof(float));
  float* states  = (float*)alloc((size_t)BATCH * NCHK * NH * PD * NST * sizeof(float));
  float* cdec    = (float*)alloc((size_t)BATCH * NH * NCHK * sizeof(float));
  float* prev    = (float*)alloc((size_t)BATCH * NCHK * NH * PD * NST * sizeof(float));
  float* Yb      = (float*)alloc((size_t)MTOK * IDIM * sizeof(float));
  bf16*  y_b     = (bf16*)alloc((size_t)MTOK * IDIM * sizeof(bf16));

  // 1. weight + activation conversion
  cvt_bf16_kernel<<<512, 256, 0, stream>>>(win, w_in_b, (long)PROJD * HDIM);
  cvt_bf16_kernel<<<512, 256, 0, stream>>>(wout, w_out_b, (long)HDIM * IDIM);
  rmsnorm_in_kernel<<<MTOK, 256, 0, stream>>>(hs, lnw, xn_b);

  // 2. in_proj GEMM: proj[M, PROJD] = xn @ W_in^T
  gemm_wave_kernel<<<dim3(PROJD / 64, MTOK / 256), 256, 0, stream>>>(
      xn_b, HDIM, w_in_b, HDIM, proj, PROJD, HDIM, nullptr);

  // 3. depthwise causal conv + SiLU
  conv_silu_kernel<<<dim3((CONVD + 255) / 256, TLEN, BATCH), 256, 0, stream>>>(
      proj, cw, cb, xbc);

  // 4. per-chunk SSD (WMMA out of LDS; ~290 KB per workgroup)
  size_t smem = (size_t)(5 * CHK * PD + CHK * CHK) * sizeof(bf16) +
                (size_t)(2 * CHK) * sizeof(float);
  ssd_chunk_kernel<<<dim3(NH, NCHK, BATCH), 256, smem, stream>>>(
      xbc, proj, dtb, alog, Yb, acum, states, cdec);

  // 5. inter-chunk recurrence + off-diagonal contribution
  ssd_scan_kernel<<<dim3(NH, BATCH), 256, 0, stream>>>(states, cdec, prev);
  ssd_off_kernel<<<dim3(NH, NCHK, BATCH), 256, 0, stream>>>(xbc, acum, prev,
                                                            Dp, Yb);

  // 6. gated RMSNorm -> bf16
  gated_norm_kernel<<<MTOK, 256, 0, stream>>>(Yb, proj, nw, y_b);

  // 7. out_proj GEMM with fused residual: out = hs + y @ W_out^T
  gemm_wave_kernel<<<dim3(HDIM / 64, MTOK / 256), 256, 0, stream>>>(
      y_b, IDIM, w_out_b, IDIM, out, HDIM, IDIM, hs);
}
